// VNPointNetEncoder_68779606278266
// MI455X (gfx1250) — matrix operand order repaired and checked by reading, not compiled, not run on hardware
//
#include <hip/hip_runtime.h>

typedef __attribute__((ext_vector_type(16))) _Float16 v16h;
typedef __attribute__((ext_vector_type(8)))  _Float16 v8h;
typedef __attribute__((ext_vector_type(8)))  float    v8f;

#define EPSV  1e-6f
#define BNEPS 1e-5f

constexpr int B = 16, N = 2048, KN = 20;
constexpr int NCOL = 3 * N;  // 6144 columns per batch per channel

// ---------------------------------------------------------------- kNN ------
__global__ __launch_bounds__(256) void knn_kernel(const float* __restrict__ x,
                                                  int* __restrict__ idx) {
  __shared__ float xs[3 * N];
  __shared__ float xx[N];
  const int b = blockIdx.x;
  const int tid = threadIdx.x;
  for (int t = tid; t < 3 * N; t += 256) xs[t] = x[(size_t)b * 3 * N + t];
  __syncthreads();
  for (int m = tid; m < N; m += 256) {
    float a0 = xs[m], a1 = xs[N + m], a2 = xs[2 * N + m];
    xx[m] = a0 * a0 + a1 * a1 + a2 * a2;
  }
  __syncthreads();
  const int wave = tid >> 5, lane = tid & 31;
  const int n = blockIdx.y * 8 + wave;
  const float c0 = xs[n], c1 = xs[N + n], c2 = xs[2 * N + n];
  const float xxn = xx[n];
  float v[KN]; int id[KN];
#pragma unroll
  for (int t = 0; t < KN; ++t) { v[t] = -3.0e38f; id[t] = 0; }
  for (int m = lane; m < N; m += 32) {
    float dot = c0 * xs[m] + c1 * xs[N + m] + c2 * xs[2 * N + m];
    float pd = 2.f * dot - xxn - xx[m];
    if (pd > v[KN - 1]) {
      v[KN - 1] = pd; id[KN - 1] = m;
#pragma unroll
      for (int t = KN - 1; t > 0; --t) {
        if (v[t] > v[t - 1]) {
          float tv = v[t]; v[t] = v[t - 1]; v[t - 1] = tv;
          int ti = id[t]; id[t] = id[t - 1]; id[t - 1] = ti;
        }
      }
    }
  }
  int* out = idx + ((size_t)b * N + n) * KN;
  for (int j = 0; j < KN; ++j) {
    float bv = v[0]; int bl = lane;
#pragma unroll
    for (int off = 16; off > 0; off >>= 1) {
      float ov = __shfl_xor(bv, off, 32);
      int   ol = __shfl_xor(bl, off, 32);
      if (ov > bv || (ov == bv && ol < bl)) { bv = ov; bl = ol; }
    }
    int widx = __shfl(id[0], bl, 32);
    if (lane == 0) out[j] = widx;
    if (lane == bl) {
#pragma unroll
      for (int t = 0; t < KN - 1; ++t) { v[t] = v[t + 1]; id[t] = id[t + 1]; }
      v[KN - 1] = -3.0e38f;
    }
  }
}

// ------------------------------------------------------ edge feature -------
__device__ inline void edge_vec(const float* __restrict__ xb, int n, int j, float* e) {
  float cx = xb[n], cy = xb[N + n], cz = xb[2 * N + n];
  float ax = xb[j], ay = xb[N + j], az = xb[2 * N + j];
  e[0] = ax - cx; e[1] = ay - cy; e[2] = az - cz;   // nb - ctr
  e[3] = cx;      e[4] = cy;      e[5] = cz;        // ctr
  e[6] = ay * cz - az * cy;                         // cross(nb, ctr)
  e[7] = az * cx - ax * cz;
  e[8] = ax * cy - ay * cx;
}

// stats pass: deterministic block-tree partials of norm / norm^2 per channel
__global__ __launch_bounds__(256) void edge_stats_kernel(const float* __restrict__ x,
                                                         const int* __restrict__ idx,
                                                         const float* __restrict__ Wp,
                                                         float* __restrict__ part) {
  __shared__ float sm[16 * 256];
  const int tid = threadIdx.x;
  const int e = blockIdx.x * 256 + tid;
  const int b = e / (N * KN);
  const int r = e - b * (N * KN);
  const int n = r / KN;
  const float* xb = x + (size_t)b * 3 * N;
  int j = idx[(size_t)e];
  float ev[9];
  edge_vec(xb, n, j, ev);
  float nr[16];
#pragma unroll
  for (int o = 0; o < 16; ++o) {
    float w0 = Wp[o * 3], w1 = Wp[o * 3 + 1], w2 = Wp[o * 3 + 2];
    float p0 = w0 * ev[0] + w1 * ev[3] + w2 * ev[6];
    float p1 = w0 * ev[1] + w1 * ev[4] + w2 * ev[7];
    float p2 = w0 * ev[2] + w1 * ev[5] + w2 * ev[8];
    nr[o] = sqrtf(p0 * p0 + p1 * p1 + p2 * p2) + EPSV;
  }
#pragma unroll
  for (int o = 0; o < 16; ++o) sm[o * 256 + tid] = nr[o];
  __syncthreads();
  for (int st = 128; st > 0; st >>= 1) {
    if (tid < st)
#pragma unroll
      for (int o = 0; o < 16; ++o) sm[o * 256 + tid] += sm[o * 256 + tid + st];
    __syncthreads();
  }
  if (tid < 16) part[tid * gridDim.x + blockIdx.x] = sm[tid * 256];
  __syncthreads();
#pragma unroll
  for (int o = 0; o < 16; ++o) sm[o * 256 + tid] = nr[o] * nr[o];
  __syncthreads();
  for (int st = 128; st > 0; st >>= 1) {
    if (tid < st)
#pragma unroll
      for (int o = 0; o < 16; ++o) sm[o * 256 + tid] += sm[o * 256 + tid + st];
    __syncthreads();
  }
  if (tid < 16) part[(16 + tid) * gridDim.x + blockIdx.x] = sm[tid * 256];
}

__global__ __launch_bounds__(256) void edge_stats_final_kernel(const float* __restrict__ part,
                                                               float* __restrict__ stats) {
  __shared__ float s1[256], s2[256];
  const int o = blockIdx.x, tid = threadIdx.x;
  float a = 0.f, q = 0.f;
  for (int t = tid; t < 2560; t += 256) {
    a += part[o * 2560 + t];
    q += part[(16 + o) * 2560 + t];
  }
  s1[tid] = a; s2[tid] = q;
  __syncthreads();
  for (int st = 128; st > 0; st >>= 1) {
    if (tid < st) { s1[tid] += s1[tid + st]; s2[tid] += s2[tid + st]; }
    __syncthreads();
  }
  if (tid == 0) { stats[o * 2] = s1[0]; stats[o * 2 + 1] = s2[0]; }
}

// apply pass: recompute edge features, BN + directional lrelu, mean over K
__global__ __launch_bounds__(64) void edge_apply_kernel(const float* __restrict__ x,
                                                        const int* __restrict__ idx,
                                                        const float* __restrict__ Wp,
                                                        const float* __restrict__ Wdp,
                                                        const float* __restrict__ g,
                                                        const float* __restrict__ bt,
                                                        const float* __restrict__ stats,
                                                        float* __restrict__ h1) {
  __shared__ float ev[KN][9];
  __shared__ float red[16][4][3];
  const int bn = blockIdx.x;
  const int b = bn >> 11, n = bn & 2047;
  const int tid = threadIdx.x;
  const float* xb = x + (size_t)b * 3 * N;
  if (tid < KN) {
    int j = idx[(size_t)bn * KN + tid];
    edge_vec(xb, n, j, ev[tid]);
  }
  __syncthreads();
  const int o = tid & 15, ks = tid >> 4;
  const float cntInv = 1.0f / (float)((size_t)B * N * KN);
  float mu  = stats[o * 2] * cntInv;
  float var = stats[o * 2 + 1] * cntInv - mu * mu;
  float rs  = rsqrtf(var + BNEPS);
  float ga = g[o], be = bt[o];
  float w0 = Wp[o * 3],  w1 = Wp[o * 3 + 1],  w2 = Wp[o * 3 + 2];
  float u0 = Wdp[o * 3], u1 = Wdp[o * 3 + 1], u2 = Wdp[o * 3 + 2];
  float a0 = 0.f, a1 = 0.f, a2 = 0.f;
  for (int k = ks; k < KN; k += 4) {
    const float* e = ev[k];
    float p0 = w0 * e[0] + w1 * e[3] + w2 * e[6];
    float p1 = w0 * e[1] + w1 * e[4] + w2 * e[7];
    float p2 = w0 * e[2] + w1 * e[5] + w2 * e[8];
    float d0 = u0 * e[0] + u1 * e[3] + u2 * e[6];
    float d1 = u0 * e[1] + u1 * e[4] + u2 * e[7];
    float d2 = u0 * e[2] + u1 * e[5] + u2 * e[8];
    float nrm = sqrtf(p0 * p0 + p1 * p1 + p2 * p2) + EPSV;
    float sc = (ga * (nrm - mu) * rs + be) / nrm;
    p0 *= sc; p1 *= sc; p2 *= sc;
    float dot = p0 * d0 + p1 * d1 + p2 * d2;
    if (dot < 0.f) {
      float f = dot / (d0 * d0 + d1 * d1 + d2 * d2 + EPSV);
      p0 -= f * d0; p1 -= f * d1; p2 -= f * d2;
    }
    a0 += p0; a1 += p1; a2 += p2;
  }
  red[o][ks][0] = a0; red[o][ks][1] = a1; red[o][ks][2] = a2;
  __syncthreads();
  if (ks == 0) {
    float s0 = red[o][0][0] + red[o][1][0] + red[o][2][0] + red[o][3][0];
    float s1 = red[o][0][1] + red[o][1][1] + red[o][2][1] + red[o][3][1];
    float s2 = red[o][0][2] + red[o][1][2] + red[o][2][2] + red[o][3][2];
    float* out = h1 + ((size_t)(b * 16 + o) * 3) * N + n;
    const float inv = 1.0f / (float)KN;
    out[0] = s0 * inv; out[N] = s1 * inv; out[2 * N] = s2 * inv;
  }
}

// ------------------------------------ WMMA channel-mixing GEMM -------------
// Y[b] (O x 6144) = W (O x I) @ X[b] (I x 6144), f16 inputs, f32 accumulate.
// LDS staged so each lane's WMMA fragment is two contiguous ds_load_b128:
//   A 16x32 f16 mapping per lane: K in {8h..8h+7} U {16+8h..16+8h+7}  (2 runs)
//   B 32x16 f16 mapping per lane: K in {8h..8h+7} U {16+8h..16+8h+7}  (2 runs) -> K-major tile
__global__ __launch_bounds__(256) void gemm_vn_kernel(const float* __restrict__ Wm,
                                                      const float* __restrict__ X,
                                                      float* __restrict__ Y,
                                                      int O, int I) {
  __shared__ _Float16 Wt[16 * 136];    // row-major, pitch Ipad+8 (bank-rotated)
  __shared__ _Float16 XtT[128 * 136];  // K-major (transposed), pitch Ipad+8
  const int tid = threadIdx.x;
  const int Ipad = (I + 31) & ~31;
  const int pitch = Ipad + 8;          // 16B-aligned rows, 4-dword bank rotation
  const int rowbase = blockIdx.y * 16;
  const int colbase = blockIdx.x * 128;
  const float* Xb = X + (size_t)blockIdx.z * I * NCOL;
  for (int t = tid; t < 16 * Ipad; t += 256) {
    int rr = t / Ipad, ii = t - rr * Ipad;
    Wt[rr * pitch + ii] = (ii < I) ? (_Float16)Wm[(rowbase + rr) * I + ii]
                                   : (_Float16)0.f;
  }
  for (int t = tid; t < Ipad * 128; t += 256) {
    int ii = t >> 7, cc = t & 127;   // ii = K (channel), cc = column
    float val = 0.f;
    if (ii < I) {
      val = Xb[(size_t)ii * NCOL + colbase + cc];
      if (cc == 0 && colbase + 128 < NCOL)
        __builtin_prefetch(&Xb[(size_t)ii * NCOL + colbase + 128], 0, 1);
    }
    XtT[cc * pitch + ii] = (_Float16)val;   // transpose on store
  }
  __syncthreads();
  const int wave = tid >> 5, lane = tid & 31;
  const int half = lane >> 4, mn = lane & 15;
  const int col = wave * 16 + mn;
  v8f acc;
#pragma unroll
  for (int q = 0; q < 8; ++q) acc[q] = 0.f;
  for (int k0 = 0; k0 < Ipad; k0 += 32) {
    // A fragment: two contiguous 8-half runs from the weight tile row
    const v8h* wp = (const v8h*)&Wt[mn * pitch + k0 + 8 * half];
    v8h a_lo = wp[0];   // K = k0 + 8*half + [0..7]
    v8h a_hi = wp[2];   // K = k0 + 16 + 8*half + [0..7]
    v16h av = __builtin_shufflevector(a_lo, a_hi, 0, 1, 2, 3, 4, 5, 6, 7,
                                      8, 9, 10, 11, 12, 13, 14, 15);
    // B fragment: two contiguous 8-half runs from the K-major activation tile
    const v8h* xp = (const v8h*)&XtT[col * pitch + k0 + 8 * half];
    v16h bv = __builtin_shufflevector(xp[0], xp[2], 0, 1, 2, 3, 4, 5, 6, 7,
                                      8, 9, 10, 11, 12, 13, 14, 15);
    acc = __builtin_amdgcn_wmma_f32_16x16x32_f16(false, av, false, bv,
                                                 (short)0, acc, false, false);
  }
  float* Yb = Y + (size_t)blockIdx.z * O * NCOL;
#pragma unroll
  for (int q = 0; q < 8; ++q) {
    int row = rowbase + q + 8 * half;  // D layout: VGPR q -> M = q + 8*(lane/16)
    Yb[(size_t)row * NCOL + colbase + col] = acc[q];
  }
}

// ------------------------------------------- BN stats (per channel) --------
__global__ __launch_bounds__(256) void norm_stats_kernel(const float* __restrict__ P,
                                                         int O, float* __restrict__ part) {
  __shared__ float s1[256], s2[256];
  const int tid = threadIdx.x, o = blockIdx.y;
  const int s = blockIdx.x * 256 + tid;
  const int b = s >> 11, n = s & 2047;
  size_t base = ((size_t)(b * O + o) * 3) * N + n;
  float p0 = P[base], p1 = P[base + N], p2 = P[base + 2 * N];
  float nrm = sqrtf(p0 * p0 + p1 * p1 + p2 * p2) + EPSV;
  s1[tid] = nrm; s2[tid] = nrm * nrm;
  __syncthreads();
  for (int st = 128; st > 0; st >>= 1) {
    if (tid < st) { s1[tid] += s1[tid + st]; s2[tid] += s2[tid + st]; }
    __syncthreads();
  }
  if (tid == 0) {
    part[o * 128 + blockIdx.x]       = s1[0];
    part[(O + o) * 128 + blockIdx.x] = s2[0];
  }
}

__global__ __launch_bounds__(128) void stats_final_kernel(const float* __restrict__ part,
                                                          int O, float* __restrict__ stats) {
  __shared__ float s1[128], s2[128];
  const int o = blockIdx.x, tid = threadIdx.x;
  s1[tid] = part[o * 128 + tid];
  s2[tid] = part[(O + o) * 128 + tid];
  __syncthreads();
  for (int st = 64; st > 0; st >>= 1) {
    if (tid < st) { s1[tid] += s1[tid + st]; s2[tid] += s2[tid + st]; }
    __syncthreads();
  }
  if (tid == 0) { stats[o * 2] = s1[0]; stats[o * 2 + 1] = s2[0]; }
}

// ------------------------------- BN apply (+ optional directional lrelu) ---
__global__ __launch_bounds__(256) void bn_act_kernel(const float* __restrict__ P,
                                                     const float* __restrict__ D,
                                                     const float* __restrict__ stats,
                                                     const float* __restrict__ g,
                                                     const float* __restrict__ bt,
                                                     float* __restrict__ out, int O) {
  const int s = blockIdx.x * 256 + threadIdx.x;
  const int n = s & 2047;
  const int t = s >> 11;
  const int o = t % O, b = t / O;
  const float cntInv = 1.0f / (float)((size_t)B * N);
  float mu  = stats[o * 2] * cntInv;
  float var = stats[o * 2 + 1] * cntInv - mu * mu;
  float rs  = rsqrtf(var + BNEPS);
  size_t base = ((size_t)(b * O + o) * 3) * N + n;
  float p0 = P[base], p1 = P[base + N], p2 = P[base + 2 * N];
  float nrm = sqrtf(p0 * p0 + p1 * p1 + p2 * p2) + EPSV;
  float sc = (g[o] * (nrm - mu) * rs + bt[o]) / nrm;
  p0 *= sc; p1 *= sc; p2 *= sc;
  if (D) {
    float d0 = D[base], d1 = D[base + N], d2 = D[base + 2 * N];
    float dot = p0 * d0 + p1 * d1 + p2 * d2;
    if (dot < 0.f) {
      float f = dot / (d0 * d0 + d1 * d1 + d2 * d2 + EPSV);
      p0 -= f * d0; p1 -= f * d1; p2 -= f * d2;
    }
  }
  out[base] = p0; out[base + N] = p1; out[base + 2 * N] = p2;
}

// --------------------------------------------- mean over N, concat ---------
__global__ __launch_bounds__(256) void mean_n_kernel(const float* __restrict__ H,
                                                     float* __restrict__ means) {
  __shared__ float s1[256];
  const int r = blockIdx.x, tid = threadIdx.x;
  const float* p = H + (size_t)r * N;
  float a = 0.f;
  for (int t = tid; t < N; t += 256) a += p[t];
  s1[tid] = a;
  __syncthreads();
  for (int st = 128; st > 0; st >>= 1) {
    if (tid < st) s1[tid] += s1[tid + st];
    __syncthreads();
  }
  if (tid == 0) means[r] = s1[0] * (1.0f / (float)N);
}

__global__ __launch_bounds__(256) void concat_kernel(const float* __restrict__ H,
                                                     const float* __restrict__ means,
                                                     float* __restrict__ out, int C) {
  const int s = blockIdx.x * 256 + threadIdx.x;
  const int n = s & 2047;
  int t = s >> 11;
  const int v = t % 3; t /= 3;
  const int c = t % (2 * C);
  const int b = t / (2 * C);
  float val;
  if (c < C) val = H[((size_t)(b * C + c) * 3 + v) * N + n];
  else       val = means[(b * C + (c - C)) * 3 + v];
  out[(size_t)s] = val;
}

// ------------------------------------------- z0 = W_lin (3x32) @ z2 --------
__global__ __launch_bounds__(256) void zlin_kernel(const float* __restrict__ Z,
                                                   const float* __restrict__ WL,
                                                   float* __restrict__ z0) {
  __shared__ float wl[96];
  const int tid = threadIdx.x;
  if (tid < 96) wl[tid] = WL[tid];
  __syncthreads();
  const int s = blockIdx.x * 256 + tid;
  const int n = s & 2047;
  const int t = s >> 11;
  const int v = t % 3, b = t / 3;
  float zi[32];
#pragma unroll
  for (int i = 0; i < 32; ++i) zi[i] = Z[((size_t)(b * 32 + i) * 3 + v) * N + n];
#pragma unroll
  for (int o = 0; o < 3; ++o) {
    float a = 0.f;
#pragma unroll
    for (int i = 0; i < 32; ++i) a += wl[o * 32 + i] * zi[i];
    z0[((size_t)(b * 3 + o) * 3 + v) * N + n] = a;
  }
}

// --------------------- final per-point 128x3 . 3x3 product + max over N ----
__global__ __launch_bounds__(256) void final_kernel(const float* __restrict__ Hc,
                                                    const float* __restrict__ z0,
                                                    float* __restrict__ out) {
  __shared__ float red[3][256];
  const int b = blockIdx.x, i = blockIdx.y, tid = threadIdx.x;
  float m0 = -3.0e38f, m1 = -3.0e38f, m2 = -3.0e38f;
  const float* hb = Hc + ((size_t)(b * 128 + i) * 3) * N;
  const float* zb = z0 + (size_t)b * 9 * N;
  for (int n = tid; n < N; n += 256) {
    float h0 = hb[n], h1 = hb[N + n], h2 = hb[2 * N + n];
    float a0 = h0 * zb[n]         + h1 * zb[N + n]     + h2 * zb[2 * N + n];
    float a1 = h0 * zb[3 * N + n] + h1 * zb[4 * N + n] + h2 * zb[5 * N + n];
    float a2 = h0 * zb[6 * N + n] + h1 * zb[7 * N + n] + h2 * zb[8 * N + n];
    m0 = fmaxf(m0, a0); m1 = fmaxf(m1, a1); m2 = fmaxf(m2, a2);
  }
  red[0][tid] = m0; red[1][tid] = m1; red[2][tid] = m2;
  __syncthreads();
  for (int st = 128; st > 0; st >>= 1) {
    if (tid < st) {
      red[0][tid] = fmaxf(red[0][tid], red[0][tid + st]);
      red[1][tid] = fmaxf(red[1][tid], red[1][tid + st]);
      red[2][tid] = fmaxf(red[2][tid], red[2][tid + st]);
    }
    __syncthreads();
  }
  if (tid == 0) {
    out[b * 384 + i * 3 + 0] = red[0][0];
    out[b * 384 + i * 3 + 1] = red[1][0];
    out[b * 384 + i * 3 + 2] = red[2][0];
  }
}

// ---------------------------------------------------------------- host -----
extern "C" void kernel_launch(void* const* d_in, const int* in_sizes, int n_in,
                              void* d_out, int out_size, void* d_ws, size_t ws_size,
                              hipStream_t stream) {
  const float* x     = (const float*)d_in[0];
  const float* W_pos = (const float*)d_in[1];
  const float* Wd_pos= (const float*)d_in[2];
  const float* g_pos = (const float*)d_in[3];
  const float* b_pos = (const float*)d_in[4];
  const float* W1    = (const float*)d_in[5];
  const float* Wd1   = (const float*)d_in[6];
  const float* g1    = (const float*)d_in[7];
  const float* b1    = (const float*)d_in[8];
  const float* W2    = (const float*)d_in[9];
  const float* Wd2   = (const float*)d_in[10];
  const float* g2    = (const float*)d_in[11];
  const float* b2    = (const float*)d_in[12];
  const float* W3    = (const float*)d_in[13];
  const float* g3    = (const float*)d_in[14];
  const float* b3    = (const float*)d_in[15];
  const float* Ws1   = (const float*)d_in[16];
  const float* Wds1  = (const float*)d_in[17];
  const float* gs1   = (const float*)d_in[18];
  const float* bs1   = (const float*)d_in[19];
  const float* Ws2   = (const float*)d_in[20];
  const float* Wds2  = (const float*)d_in[21];
  const float* gs2   = (const float*)d_in[22];
  const float* bs2   = (const float*)d_in[23];
  const float* W_lin = (const float*)d_in[24];
  (void)in_sizes; (void)n_in; (void)out_size; (void)ws_size;

  char* ws = (char*)d_ws;
  auto chBytes = [](int C) { return (size_t)B * C * 3 * N * sizeof(float); };
  size_t off = 0;
  int*   idx    = (int*)(ws + off);   off += (size_t)B * N * KN * sizeof(int);
  float* bufA   = (float*)(ws + off); off += chBytes(128);  // hcat2 (persists)
  float* bufB   = (float*)(ws + off); off += chBytes(64);   // h1 / hcat1 / z1
  float* bufP   = (float*)(ws + off); off += chBytes(64);
  float* bufD   = (float*)(ws + off); off += chBytes(64);
  float* bufC   = (float*)(ws + off); off += chBytes(32);   // h2 / h3 / z2
  float* z0b    = (float*)(ws + off); off += (size_t)B * 9 * N * sizeof(float);
  float* statsE = (float*)(ws + off); off += 16 * 2 * sizeof(float);
  float* statsL = (float*)(ws + off); off += 64 * 2 * sizeof(float);
  float* partE  = (float*)(ws + off); off += 32 * 2560 * sizeof(float);
  float* partL  = (float*)(ws + off); off += 128 * 128 * sizeof(float);
  float* means  = (float*)(ws + off); off += (size_t)B * 64 * 3 * sizeof(float);

  // Stage 0: kNN indices
  knn_kernel<<<dim3(B, N / 8), 256, 0, stream>>>(x, idx);

  // Stage 1: edge-feature VN layer (two passes for global BN stats)
  edge_stats_kernel<<<dim3((B * N * KN) / 256), 256, 0, stream>>>(x, idx, W_pos, partE);
  edge_stats_final_kernel<<<dim3(16), 256, 0, stream>>>(partE, statsE);
  edge_apply_kernel<<<dim3(B * N), 64, 0, stream>>>(x, idx, W_pos, Wd_pos,
                                                    g_pos, b_pos, statsE, bufB);  // h1

  auto vnlin = [&](const float* Wm, const float* Xin, float* Yout, int O, int I) {
    gemm_vn_kernel<<<dim3(NCOL / 128, O / 16, B), 256, 0, stream>>>(Wm, Xin, Yout, O, I);
  };
  auto vnstats = [&](const float* Pb, int O) {
    norm_stats_kernel<<<dim3((B * N) / 256, O), 256, 0, stream>>>(Pb, O, partL);
    stats_final_kernel<<<dim3(O), 128, 0, stream>>>(partL, O, statsL);
  };
  auto vnlrelu = [&](const float* Xin, const float* Wm, const float* Wdm,
                     const float* gm, const float* bm, int O, int I, float* Out) {
    vnlin(Wm, Xin, bufP, O, I);
    vnlin(Wdm, Xin, bufD, O, I);
    vnstats(bufP, O);
    bn_act_kernel<<<dim3((B * O * N) / 256), 256, 0, stream>>>(bufP, bufD, statsL,
                                                               gm, bm, Out, O);
  };

  // Layer 1: 16 -> 16
  vnlrelu(bufB, W1, Wd1, g1, b1, 16, 16, bufC);                          // h2
  mean_n_kernel<<<dim3(B * 16 * 3), 256, 0, stream>>>(bufC, means);
  concat_kernel<<<dim3((B * 32 * 3 * N) / 256), 256, 0, stream>>>(bufC, means, bufB, 16); // hcat1
  // Layer 2: 32 -> 32
  vnlrelu(bufB, W2, Wd2, g2, b2, 32, 32, bufC);                          // h3
  // Layer 3: 32 -> 64, BN only
  vnlin(W3, bufC, bufP, 64, 32);
  vnstats(bufP, 64);
  bn_act_kernel<<<dim3((B * 64 * N) / 256), 256, 0, stream>>>(bufP, nullptr, statsL,
                                                              g3, b3, bufP, 64);  // h4 in-place
  mean_n_kernel<<<dim3(B * 64 * 3), 256, 0, stream>>>(bufP, means);
  concat_kernel<<<dim3((B * 128 * 3 * N) / 256), 256, 0, stream>>>(bufP, means, bufA, 64); // hcat2
  // Std layers: 128 -> 64 -> 32
  vnlrelu(bufA, Ws1, Wds1, gs1, bs1, 64, 128, bufB);                     // z1
  vnlrelu(bufB, Ws2, Wds2, gs2, bs2, 32, 64, bufC);                      // z2
  // z0 and final einsum + max pool
  zlin_kernel<<<dim3((B * 3 * N) / 256), 256, 0, stream>>>(bufC, W_lin, z0b);
  final_kernel<<<dim3(B, 128), 256, 0, stream>>>(bufA, z0b, (float*)d_out);
}
